// MoEFFN_69836168233264
// MI455X (gfx1250) — matrix operand order repaired and checked by reading, not compiled, not run on hardware
//
#include <hip/hip_runtime.h>
#include <math.h>

// Problem constants (from reference): B=2, L=2048 -> T=4096
#define T_TOK 4096
#define DDIM  1024
#define FDIM  2048
#define NE    8
#define NTOPK 2
#define TK    (T_TOK * NTOPK)      // 8192 compact rows

#define KSTEP 32                   // bf16 WMMA K
#define KPAD  40                   // LDS row stride (ushort) -> 80B, 16B aligned
#define TM    128

typedef __attribute__((ext_vector_type(16))) __bf16 v16bf;
typedef __attribute__((ext_vector_type(8)))  float  v8f;
typedef unsigned int u32x4 __attribute__((ext_vector_type(4)));

union Frag16 { v16bf v; u32x4 q[2]; };

__device__ __forceinline__ unsigned short f2bf(float x) {
    unsigned int b = __float_as_uint(x);
    b += 0x7FFFu + ((b >> 16) & 1u);          // round to nearest even
    return (unsigned short)(b >> 16);
}

// ---------------------------------------------------------------------------
// x f32 -> bf16 (4 elems / thread)
// ---------------------------------------------------------------------------
__global__ void k_cvt_x(const float* __restrict__ x, unsigned short* __restrict__ xbf) {
    size_t i = (size_t)blockIdx.x * blockDim.x + threadIdx.x;
    float4 v = ((const float4*)x)[i];
    ushort4 o;
    o.x = f2bf(v.x); o.y = f2bf(v.y); o.z = f2bf(v.z); o.w = f2bf(v.w);
    ((ushort4*)xbf)[i] = o;
}

// ---------------------------------------------------------------------------
// Per-expert transpose + f32->bf16:  in [R][C] f32  ->  out [C][R] bf16
// grid: (C/32, R/32, E), block 256
// ---------------------------------------------------------------------------
__global__ void k_transpose_bf16(const float* __restrict__ in,
                                 unsigned short* __restrict__ out,
                                 int R, int C) {
    __shared__ float tile[32][33];
    size_t mat = (size_t)R * C;
    in  += (size_t)blockIdx.z * mat;
    out += (size_t)blockIdx.z * mat;
    int rb = blockIdx.y * 32, cb = blockIdx.x * 32;
    int t = threadIdx.x;
    int c = t & 31;
#pragma unroll
    for (int i = 0; i < 4; i++) {
        int r = (t >> 5) + i * 8;
        tile[r][c] = in[(size_t)(rb + r) * C + cb + c];
    }
    __syncthreads();
    int r2 = t & 31;
#pragma unroll
    for (int i = 0; i < 4; i++) {
        int c2 = (t >> 5) + i * 8;
        out[(size_t)(cb + c2) * R + rb + r2] = f2bf(tile[r2][c2]);
    }
}

// ---------------------------------------------------------------------------
// Router: one wave32 per token -> logits(8), softmax, top-2, renorm
// ---------------------------------------------------------------------------
__global__ __launch_bounds__(256)
void k_gate(const float* __restrict__ x, const float* __restrict__ Wg,
            const float* __restrict__ bias,
            int* __restrict__ topIdx, float* __restrict__ topW,
            int* __restrict__ counts) {
    int wid = threadIdx.x >> 5, lane = threadIdx.x & 31;
    int t = blockIdx.x * 8 + wid;
    float acc[NE];
#pragma unroll
    for (int e = 0; e < NE; e++) acc[e] = 0.0f;
    const float* xr = x + (size_t)t * DDIM;
    for (int d = lane; d < DDIM; d += 32) {
        float xv = xr[d];
        const float4* wg = (const float4*)(Wg + (size_t)d * NE);
        float4 w0 = wg[0], w1 = wg[1];
        acc[0] += xv * w0.x; acc[1] += xv * w0.y; acc[2] += xv * w0.z; acc[3] += xv * w0.w;
        acc[4] += xv * w1.x; acc[5] += xv * w1.y; acc[6] += xv * w1.z; acc[7] += xv * w1.w;
    }
#pragma unroll
    for (int off = 16; off > 0; off >>= 1)
#pragma unroll
        for (int e = 0; e < NE; e++) acc[e] += __shfl_down(acc[e], off, 32);

    if (lane == 0) {
        float l[NE], p[NE];
        float m = -1e30f;
#pragma unroll
        for (int e = 0; e < NE; e++) { l[e] = acc[e] + bias[e]; m = fmaxf(m, l[e]); }
#pragma unroll
        for (int e = 0; e < NE; e++) p[e] = __expf(l[e] - m);
        int i0 = 0; float p0 = p[0];
#pragma unroll
        for (int e = 1; e < NE; e++) if (p[e] > p0) { p0 = p[e]; i0 = e; }
        int i1 = -1; float p1 = -1.0f;
#pragma unroll
        for (int e = 0; e < NE; e++) if (e != i0 && p[e] > p1) { p1 = p[e]; i1 = e; }
        float inv = 1.0f / (p0 + p1);
        topIdx[t * 2 + 0] = i0; topW[t * 2 + 0] = p0 * inv;
        topIdx[t * 2 + 1] = i1; topW[t * 2 + 1] = p1 * inv;
        atomicAdd(&counts[i0], 1);
        atomicAdd(&counts[i1], 1);
    }
}

__global__ void k_prefix(const int* __restrict__ counts, int* __restrict__ prefix,
                         float* __restrict__ outCounts) {
    if (threadIdx.x == 0) {
        int s = 0;
        for (int e = 0; e < NE; e++) { prefix[e] = s; s += counts[e]; outCounts[e] = (float)counts[e]; }
    }
}

__global__ void k_assign(const int* __restrict__ topIdx, const float* __restrict__ topW,
                         const int* __restrict__ prefix, int* __restrict__ cursor,
                         int* __restrict__ rowTok, float* __restrict__ rowW) {
    int t = blockIdx.x * blockDim.x + threadIdx.x;
    if (t >= T_TOK) return;
#pragma unroll
    for (int k = 0; k < NTOPK; k++) {
        int e = topIdx[t * 2 + k];
        int pos = atomicAdd(&cursor[e], 1);
        int row = prefix[e] + pos;
        rowTok[row] = t;
        rowW[row]   = topW[t * 2 + k];
    }
}

// ---------------------------------------------------------------------------
// GEMM1 + fused SwiGLU:  a[row, n] = silu(x@W1u) * (x@W1v), bf16 out
// A: gathered token rows (xbf) [TM=128], B: W1t [2F][D] bf16, N-tile 64(u)+64(v)
// Waves: 4(M) x 2(N); each wave: 2 A-frags, 2 u + 2 v B-frags, 8 WMMAs / K-step
// Double-buffered LDS, one barrier per K-step.   grid (32, F/64, E)
// ---------------------------------------------------------------------------
__global__ __launch_bounds__(256)
void k_gemm1(const unsigned short* __restrict__ xbf,
             const unsigned short* __restrict__ W1t,
             const int* __restrict__ rowTok,
             const int* __restrict__ counts, const int* __restrict__ prefix,
             unsigned short* __restrict__ abuf) {
    const int e = blockIdx.z;
    const int cnt = counts[e];
    const int mBase = blockIdx.x * TM;
    if (mBase >= cnt) return;
    const int rowsRem = min(cnt - mBase, TM);
    const int rowBase = prefix[e];
    const int nb = blockIdx.y * 64;

    __shared__ unsigned short As[2][TM * KPAD];
    __shared__ unsigned short Bs[2][128 * KPAD];

    const int tid = threadIdx.x;
    const int wid = tid >> 5, lane = tid & 31;
    const int widM = wid & 3, widN = wid >> 2;

    // staging: 2 threads per row (A: 128 rows, B: 128 n-rows), 16 bf16 each
    const int sr = tid >> 1, sp = tid & 1;
    const bool aval = sr < rowsRem;
    int tokA = aval ? rowTok[rowBase + mBase + sr] : 0;
    const unsigned short* aSrc = xbf + (size_t)tokA * DDIM + sp * 16;
    const int globN = (sr < 64) ? (nb + sr) : (FDIM + nb + (sr - 64));
    const unsigned short* bSrc = W1t + ((size_t)e * (2 * FDIM) + globN) * DDIM + sp * 16;
    const int stOff = sr * KPAD + sp * 16;

    v8f accU[2][2] = {};
    v8f accV[2][2] = {};
    const int r16 = lane & 15, half = lane >> 4;

    // prologue: stage 0
    {
        u32x4 a0 = (u32x4)0, a1 = (u32x4)0;
        if (aval) { a0 = *(const u32x4*)(aSrc); a1 = *(const u32x4*)(aSrc + 8); }
        u32x4 b0 = *(const u32x4*)(bSrc);
        u32x4 b1 = *(const u32x4*)(bSrc + 8);
        *(u32x4*)&As[0][stOff] = a0; *(u32x4*)&As[0][stOff + 8] = a1;
        *(u32x4*)&Bs[0][stOff] = b0; *(u32x4*)&Bs[0][stOff + 8] = b1;
    }
    __syncthreads();

    int s = 0;
    for (int k0 = 0; k0 < DDIM; k0 += KSTEP) {
        const bool more = (k0 + KSTEP) < DDIM;
        u32x4 na0 = (u32x4)0, na1 = (u32x4)0, nb0 = (u32x4)0, nb1 = (u32x4)0;
        if (more) {
            const int kn = k0 + KSTEP;
            if (aval) { na0 = *(const u32x4*)(aSrc + kn); na1 = *(const u32x4*)(aSrc + kn + 8); }
            nb0 = *(const u32x4*)(bSrc + kn);
            nb1 = *(const u32x4*)(bSrc + kn + 8);
            if (kn + KSTEP < DDIM) __builtin_prefetch(bSrc + kn + KSTEP, 0, 0);
        }

        // load all fragments up-front (distinct registers), then 8 WMMAs
        const unsigned short* asp = &As[s][0];
        const unsigned short* bsp = &Bs[s][0];
        Frag16 af[2], bu[2], bv[2];
#pragma unroll
        for (int mi = 0; mi < 2; mi++) {
            int row = widM * 32 + mi * 16 + r16;
            af[mi].q[0] = *(const u32x4*)(&asp[row * KPAD + half * 16]);
            af[mi].q[1] = *(const u32x4*)(&asp[row * KPAD + half * 16 + 8]);
        }
#pragma unroll
        for (int ni = 0; ni < 2; ni++) {
            int bru = widN * 32 + ni * 16 + r16;
            bu[ni].q[0] = *(const u32x4*)(&bsp[bru * KPAD + half * 16]);
            bu[ni].q[1] = *(const u32x4*)(&bsp[bru * KPAD + half * 16 + 8]);
            int brv = 64 + widN * 32 + ni * 16 + r16;
            bv[ni].q[0] = *(const u32x4*)(&bsp[brv * KPAD + half * 16]);
            bv[ni].q[1] = *(const u32x4*)(&bsp[brv * KPAD + half * 16 + 8]);
        }
#pragma unroll
        for (int mi = 0; mi < 2; mi++)
#pragma unroll
            for (int ni = 0; ni < 2; ni++) {
                accU[mi][ni] = __builtin_amdgcn_wmma_f32_16x16x32_bf16(
                    false, af[mi].v, false, bu[ni].v, (short)0, accU[mi][ni], false, false);
                accV[mi][ni] = __builtin_amdgcn_wmma_f32_16x16x32_bf16(
                    false, af[mi].v, false, bv[ni].v, (short)0, accV[mi][ni], false, false);
            }

        if (more) {
            *(u32x4*)&As[s ^ 1][stOff] = na0; *(u32x4*)&As[s ^ 1][stOff + 8] = na1;
            *(u32x4*)&Bs[s ^ 1][stOff] = nb0; *(u32x4*)&Bs[s ^ 1][stOff + 8] = nb1;
        }
        __syncthreads();
        s ^= 1;
    }

    // epilogue: silu(u) * v -> bf16 abuf (compact rows)
#pragma unroll
    for (int mi = 0; mi < 2; mi++)
#pragma unroll
        for (int v = 0; v < 8; v++) {
            int mrow = widM * 32 + mi * 16 + half * 8 + v;
            if (mrow < rowsRem) {
                size_t rbase = (size_t)(rowBase + mBase + mrow) * FDIM + nb + widN * 32;
#pragma unroll
                for (int ni = 0; ni < 2; ni++) {
                    float u  = accU[mi][ni][v];
                    float vv = accV[mi][ni][v];
                    float sg = u / (1.0f + __expf(-u));
                    abuf[rbase + ni * 16 + r16] = f2bf(sg * vv);
                }
            }
        }
}

// ---------------------------------------------------------------------------
// GEMM2: y = a @ W2, scaled by gate weight, atomic-added into out
// A: abuf compact rows [*, F] bf16, B: W2t [E][D][F] bf16, N-tile 128.
// Waves: 4(M) x 2(N); wave = 32 rows x 64 cols = 8 WMMAs / K-step.
// grid (32, D/128, E)
// ---------------------------------------------------------------------------
__global__ __launch_bounds__(256)
void k_gemm2(const unsigned short* __restrict__ abuf,
             const unsigned short* __restrict__ W2t,
             const int* __restrict__ rowTok, const float* __restrict__ rowW,
             const int* __restrict__ counts, const int* __restrict__ prefix,
             float* __restrict__ out) {
    const int e = blockIdx.z;
    const int cnt = counts[e];
    const int mBase = blockIdx.x * TM;
    if (mBase >= cnt) return;
    const int rowsRem = min(cnt - mBase, TM);
    const int rowBase = prefix[e];
    const int nbD = blockIdx.y * 128;

    __shared__ unsigned short As[2][TM * KPAD];
    __shared__ unsigned short Bs[2][128 * KPAD];

    const int tid = threadIdx.x;
    const int wid = tid >> 5, lane = tid & 31;
    const int widM = wid & 3, widN = wid >> 2;

    // staging: 2 threads per row (A: 128 rows, B: 128 n-rows), 16 bf16 each
    const int sr = tid >> 1, sp = tid & 1;
    const bool aval = sr < rowsRem;
    const unsigned short* aSrc = abuf + (size_t)(rowBase + mBase + sr) * FDIM + sp * 16;
    const unsigned short* bSrc = W2t + ((size_t)e * DDIM + nbD + sr) * FDIM + sp * 16;
    const int stOff = sr * KPAD + sp * 16;

    v8f acc[2][4] = {};
    const int r16 = lane & 15, half = lane >> 4;

    // prologue: stage 0
    {
        u32x4 a0 = (u32x4)0, a1 = (u32x4)0;
        if (aval) { a0 = *(const u32x4*)(aSrc); a1 = *(const u32x4*)(aSrc + 8); }
        u32x4 b0 = *(const u32x4*)(bSrc);
        u32x4 b1 = *(const u32x4*)(bSrc + 8);
        *(u32x4*)&As[0][stOff] = a0; *(u32x4*)&As[0][stOff + 8] = a1;
        *(u32x4*)&Bs[0][stOff] = b0; *(u32x4*)&Bs[0][stOff + 8] = b1;
    }
    __syncthreads();

    int s = 0;
    for (int k0 = 0; k0 < FDIM; k0 += KSTEP) {
        const bool more = (k0 + KSTEP) < FDIM;
        u32x4 na0 = (u32x4)0, na1 = (u32x4)0, nb0 = (u32x4)0, nb1 = (u32x4)0;
        if (more) {
            const int kn = k0 + KSTEP;
            if (aval) { na0 = *(const u32x4*)(aSrc + kn); na1 = *(const u32x4*)(aSrc + kn + 8); }
            nb0 = *(const u32x4*)(bSrc + kn);
            nb1 = *(const u32x4*)(bSrc + kn + 8);
            if (kn + KSTEP < FDIM) __builtin_prefetch(bSrc + kn + KSTEP, 0, 0);
        }

        const unsigned short* asp = &As[s][0];
        const unsigned short* bsp = &Bs[s][0];
        Frag16 af[2], bf4[4];
#pragma unroll
        for (int mi = 0; mi < 2; mi++) {
            int row = widM * 32 + mi * 16 + r16;
            af[mi].q[0] = *(const u32x4*)(&asp[row * KPAD + half * 16]);
            af[mi].q[1] = *(const u32x4*)(&asp[row * KPAD + half * 16 + 8]);
        }
#pragma unroll
        for (int ni = 0; ni < 4; ni++) {
            int br = widN * 64 + ni * 16 + r16;
            bf4[ni].q[0] = *(const u32x4*)(&bsp[br * KPAD + half * 16]);
            bf4[ni].q[1] = *(const u32x4*)(&bsp[br * KPAD + half * 16 + 8]);
        }
#pragma unroll
        for (int mi = 0; mi < 2; mi++)
#pragma unroll
            for (int ni = 0; ni < 4; ni++)
                acc[mi][ni] = __builtin_amdgcn_wmma_f32_16x16x32_bf16(
                    false, af[mi].v, false, bf4[ni].v, (short)0, acc[mi][ni], false, false);

        if (more) {
            *(u32x4*)&As[s ^ 1][stOff] = na0; *(u32x4*)&As[s ^ 1][stOff + 8] = na1;
            *(u32x4*)&Bs[s ^ 1][stOff] = nb0; *(u32x4*)&Bs[s ^ 1][stOff + 8] = nb1;
        }
        __syncthreads();
        s ^= 1;
    }

#pragma unroll
    for (int mi = 0; mi < 2; mi++)
#pragma unroll
        for (int v = 0; v < 8; v++) {
            int mrow = widM * 32 + mi * 16 + half * 8 + v;
            if (mrow < rowsRem) {
                int grow = rowBase + mBase + mrow;
                int tok = rowTok[grow];
                float w = rowW[grow];
                float* orow = out + (size_t)tok * DDIM + nbD + widN * 64;
#pragma unroll
                for (int ni = 0; ni < 4; ni++)
                    unsafeAtomicAdd(&orow[ni * 16 + r16], acc[mi][ni][v] * w);
            }
        }
}

// ---------------------------------------------------------------------------
extern "C" void kernel_launch(void* const* d_in, const int* in_sizes, int n_in,
                              void* d_out, int out_size, void* d_ws, size_t ws_size,
                              hipStream_t stream) {
    const float* x    = (const float*)d_in[0];   // [T, D]
    const float* Wg   = (const float*)d_in[1];   // [D, E]
    const float* W1   = (const float*)d_in[2];   // [E, D, 2F]
    const float* W2   = (const float*)d_in[3];   // [E, F, D]
    const float* bias = (const float*)d_in[4];   // [E]
    float* out = (float*)d_out;                  // [T*D] ++ [E]

    char* p = (char*)d_ws;
    auto carve = [&](size_t bytes) -> char* {
        char* r = p;
        p += (bytes + 255) & ~(size_t)255;
        return r;
    };
    unsigned short* xbf  = (unsigned short*)carve((size_t)T_TOK * DDIM * 2);
    unsigned short* W1t  = (unsigned short*)carve((size_t)NE * 2 * FDIM * DDIM * 2);
    unsigned short* W2t  = (unsigned short*)carve((size_t)NE * DDIM * FDIM * 2);
    unsigned short* abuf = (unsigned short*)carve((size_t)TK * FDIM * 2);
    int*   rowTok = (int*)carve((size_t)TK * 4);
    float* rowW   = (float*)carve((size_t)TK * 4);
    int*   topIdx = (int*)carve((size_t)TK * 4);
    float* topW   = (float*)carve((size_t)TK * 4);
    int*   ctrl   = (int*)carve(256);
    int* counts = ctrl;        // [8]
    int* cursor = ctrl + 8;    // [8]
    int* prefix = ctrl + 16;   // [8]

    hipMemsetAsync(d_out, 0, (size_t)(T_TOK * DDIM + NE) * sizeof(float), stream);
    hipMemsetAsync(ctrl, 0, 256, stream);

    k_cvt_x<<<(T_TOK * DDIM) / (4 * 256), 256, 0, stream>>>(x, xbf);
    k_transpose_bf16<<<dim3((2 * FDIM) / 32, DDIM / 32, NE), 256, 0, stream>>>(W1, W1t, DDIM, 2 * FDIM);
    k_transpose_bf16<<<dim3(DDIM / 32, FDIM / 32, NE), 256, 0, stream>>>(W2, W2t, FDIM, DDIM);
    k_gate<<<T_TOK / 8, 256, 0, stream>>>(x, Wg, bias, topIdx, topW, counts);
    k_prefix<<<1, 32, 0, stream>>>(counts, prefix, out + (size_t)T_TOK * DDIM);
    k_assign<<<T_TOK / 256, 256, 0, stream>>>(topIdx, topW, prefix, cursor, rowTok, rowW);
    k_gemm1<<<dim3(T_TOK / TM, FDIM / 64, NE), 256, 0, stream>>>(xbf, W1t, rowTok, counts, prefix, abuf);
    k_gemm2<<<dim3(T_TOK / TM, DDIM / 128, NE), 256, 0, stream>>>(abuf, W2t, rowTok, rowW, counts, prefix, out);
}
